// GlobalAttention_24326694765263
// MI455X (gfx1250) — matrix operand order repaired and checked by reading, not compiled, not run on hardware
//
#include <hip/hip_runtime.h>

// Problem constants (match reference)
#define BB 32
#define TT 1024
#define SS 1024
#define DD 1024

typedef __attribute__((ext_vector_type(16))) __bf16 v16bf;
typedef __attribute__((ext_vector_type(8)))  float  v8f;
typedef unsigned short u16;

// Tile config: 128x128 block tile, K-step 32, 8 waves (wave32) per block.
constexpr int BMT = 128;
constexpr int BNT = 128;
constexpr int BKT = 32;
constexpr int AST  = BKT + 8;   // LDS row stride (bf16 elems), padded
constexpr int BSTR = BKT + 8;

union FragU { uint4 u[2]; v16bf v; };

// float -> bf16, round-to-nearest-even
__device__ __forceinline__ u16 f2bf(float f) {
    union { float f; unsigned u; } x; x.f = f;
    unsigned r = x.u + 0x7FFFu + ((x.u >> 16) & 1u);
    return (u16)(r >> 16);
}

// CDNA5 async global->LDS copy, 16 bytes per lane (ASYNCcnt-tracked, no VGPR
// round trip).  LDS address = low 32 bits of the generic pointer (ISA 10.2:
// LDS aperture uses addr[31:0]).
__device__ __forceinline__ void async_cp16B(void* lds_dst, const void* g_src) {
    asm volatile("global_load_async_to_lds_b128 %0, %1, off"
                 :: "v"((unsigned)(unsigned long long)lds_dst),
                    "v"((unsigned long long)g_src)
                 : "memory");
}
__device__ __forceinline__ void wait_async0() {
    asm volatile("s_wait_asynccnt 0x0" ::: "memory");
}

// All-bf16 WMMA GEMM, f32 accumulate, double-buffered LDS, async staging.
//   C[b,m,n] = epi( sum_k A[b,m,k] * B[b,n,k] )    (BTRANS=false, B is [N,K])
//   C[b,m,n] = epi( sum_k A[b,m,k] * B[b,k,n] )    (BTRANS=true,  B is [K,N])
// Dual-A: k < Ksplit reads A0, else A1 at k-Ksplit.
// EPI: 0 = store bf16, 1 = store f32, 2 = tanh + store f32.
// C index = b*Cbs + m*Crs + n  (output permutes are stride choices).
template <bool BTRANS, int EPI>
__global__ __launch_bounds__(256)
void gemm_wmma_bf16(const u16* __restrict__ A0, const u16* __restrict__ A1,
                    int Ksplit,
                    const u16* __restrict__ Bm,
                    void* __restrict__ Cv,
                    int N, int Ktot,
                    long Abs, long Ars,   // A batch / row strides (elements)
                    long Bbs,             // B batch stride (elements)
                    long Cbs, long Crs)   // C batch / row strides (elements)
{
    __shared__ u16 ldsA[2][BMT * AST];
    __shared__ u16 ldsB[2][BNT * BSTR];

    const int tid = threadIdx.x;
    const int b   = blockIdx.z;
    const int m0  = blockIdx.y * BMT;
    const int n0  = blockIdx.x * BNT;

    const int lane = tid & 31;
    const int wave = tid >> 5;
    const int wr   = wave & 3;    // 4 wave-rows * 32 = 128 M
    const int wc   = wave >> 2;   // 2 wave-cols * 64 = 128 N
    const int r    = lane & 15;
    const int half = lane >> 4;

    const u16* Ab0 = A0 + (long)b * Abs;
    const u16* Ab1 = A1 + (long)b * Abs;
    const u16* Bb  = Bm + (long)b * Bbs;

    // Per-thread staging unit: (row, 16-elem chunk); 128 rows x 2 chunks = 256.
    const int srow = tid >> 1;
    const int scol = (tid & 1) << 4;

    auto stage = [&](int k0, int p) {
        // ---- A tile (always [M,K] bf16): two async b128 per thread
        {
            const int gk = k0 + scol;
            const u16* src = (gk < Ksplit)
                ? (Ab0 + (long)(m0 + srow) * Ars + gk)
                : (Ab1 + (long)(m0 + srow) * Ars + (gk - Ksplit));
            u16* d = &ldsA[p][srow * AST + scol];
            async_cp16B(d,     src);
            async_cp16B(d + 8, src + 8);
        }
        // ---- B tile
        if constexpr (!BTRANS) {
            const u16* src = Bb + (long)(n0 + srow) * Ktot + (k0 + scol);
            u16* d = &ldsB[p][srow * BSTR + scol];
            async_cp16B(d,     src);
            async_cp16B(d + 8, src + 8);
        } else {
            // B is [K,N]: transpose while staging (sync scatter, bf16 source)
            const int k  = tid >> 3;          // 0..31
            const int nc = (tid & 7) << 4;    // 0,16,...,112
            const uint4* src = reinterpret_cast<const uint4*>(
                Bb + (long)(k0 + k) * N + (n0 + nc));
            union { u16 h[16]; uint4 u[2]; } t;
            t.u[0] = src[0];
            t.u[1] = src[1];
            #pragma unroll
            for (int j = 0; j < 16; ++j)
                ldsB[p][(nc + j) * BSTR + k] = t.h[j];
        }
    };

    const v8f vzero = {0.f, 0.f, 0.f, 0.f, 0.f, 0.f, 0.f, 0.f};
    v8f acc[2][4];
    #pragma unroll
    for (int i = 0; i < 2; ++i)
        #pragma unroll
        for (int j = 0; j < 4; ++j) acc[i][j] = vzero;

    stage(0, 0);

    int p = 0;
    for (int k0 = 0; k0 < Ktot; k0 += BKT, p ^= 1) {
        wait_async0();        // this wave's async copies for buffer p landed
        __syncthreads();      // everyone's copies landed (ds stores waited too)

        if (k0 + BKT < Ktot)
            stage(k0 + BKT, p ^ 1);   // overlap next tile with WMMA burst

        // ---- fragments per ISA 7.12.2 wave32 layouts
        FragU af[2], bf[4];
        #pragma unroll
        for (int fm = 0; fm < 2; ++fm) {
            const u16* q = &ldsA[p][(wr * 32 + fm * 16 + r) * AST + half * 8];
            af[fm].u[0] = *reinterpret_cast<const uint4*>(q);
            af[fm].u[1] = *reinterpret_cast<const uint4*>(q + 16);
        }
        #pragma unroll
        for (int fn = 0; fn < 4; ++fn) {
            const u16* q = &ldsB[p][(wc * 64 + fn * 16 + r) * BSTR + half * 8];
            bf[fn].u[0] = *reinterpret_cast<const uint4*>(q);
            bf[fn].u[1] = *reinterpret_cast<const uint4*>(q + 16);
        }

        #pragma unroll
        for (int fm = 0; fm < 2; ++fm)
            #pragma unroll
            for (int fn = 0; fn < 4; ++fn)
                acc[fm][fn] = __builtin_amdgcn_wmma_f32_16x16x32_bf16(
                    false, af[fm].v, false, bf[fn].v,
                    (short)0, acc[fm][fn], false, false);
    }

    // ---- epilogue: C layout (ISA 7.12.2): vgpr i, lane -> row = i + half*8, col = r
    #pragma unroll
    for (int fm = 0; fm < 2; ++fm)
        #pragma unroll
        for (int fn = 0; fn < 4; ++fn)
            #pragma unroll
            for (int i = 0; i < 8; ++i) {
                const int row = m0 + wr * 32 + fm * 16 + half * 8 + i;
                const int col = n0 + wc * 64 + fn * 16 + r;
                const long cidx = (long)b * Cbs + (long)row * Crs + col;
                const float v = acc[fm][fn][i];
                if constexpr (EPI == 0)
                    ((u16*)Cv)[cidx] = f2bf(v);
                else if constexpr (EPI == 1)
                    ((float*)Cv)[cidx] = v;
                else
                    ((float*)Cv)[cidx] = tanhf(v);
            }
}

// One-shot fp32 -> bf16 pack (bandwidth bound); n multiple of 4.
__global__ __launch_bounds__(256)
void pack_bf16(const float* __restrict__ src, u16* __restrict__ dst, long n)
{
    const long i = ((long)blockIdx.x * blockDim.x + threadIdx.x) * 4;
    if (i < n) {
        float4 f = *reinterpret_cast<const float4*>(src + i);
        union { u16 h[4]; uint2 u; } o;
        o.h[0] = f2bf(f.x); o.h[1] = f2bf(f.y);
        o.h[2] = f2bf(f.z); o.h[3] = f2bf(f.w);
        *reinterpret_cast<uint2*>(dst + i) = o.u;
    }
}

// In-place row softmax over S contiguous floats ([T,B,S] rows), plus a bf16
// copy at [B,T,S] for the downstream WMMA GEMM.  blockIdx.x = t*B + b.
__global__ __launch_bounds__(256)
void softmax_rows(float* __restrict__ align, u16* __restrict__ align_bf)
{
    __shared__ float red[256];
    float* row = align + (size_t)blockIdx.x * SS;
    const int t = threadIdx.x;
    const int t_idx = blockIdx.x / BB;
    const int b_idx = blockIdx.x % BB;
    u16* brow = align_bf + ((long)b_idx * TT + t_idx) * SS;

    float4 x = reinterpret_cast<float4*>(row)[t];   // 256 * 4 = 1024
    float m = fmaxf(fmaxf(x.x, x.y), fmaxf(x.z, x.w));
    red[t] = m;
    __syncthreads();
    #pragma unroll
    for (int s = 128; s > 0; s >>= 1) {
        if (t < s) red[t] = fmaxf(red[t], red[t + s]);
        __syncthreads();
    }
    m = red[0];
    __syncthreads();

    x.x = __expf(x.x - m);
    x.y = __expf(x.y - m);
    x.z = __expf(x.z - m);
    x.w = __expf(x.w - m);
    red[t] = x.x + x.y + x.z + x.w;
    __syncthreads();
    #pragma unroll
    for (int s = 128; s > 0; s >>= 1) {
        if (t < s) red[t] += red[t + s];
        __syncthreads();
    }
    const float inv = 1.0f / red[0];
    x.x *= inv; x.y *= inv; x.z *= inv; x.w *= inv;
    reinterpret_cast<float4*>(row)[t] = x;

    union { u16 h[4]; uint2 u; } o;
    o.h[0] = f2bf(x.x); o.h[1] = f2bf(x.y);
    o.h[2] = f2bf(x.z); o.h[3] = f2bf(x.w);
    *reinterpret_cast<uint2*>(brow + 4 * t) = o.u;
}

extern "C" void kernel_launch(void* const* d_in, const int* in_sizes, int n_in,
                              void* d_out, int out_size, void* d_ws, size_t ws_size,
                              hipStream_t stream)
{
    const float* input   = (const float*)d_in[0];   // [B,T,D]
    const float* context = (const float*)d_in[1];   // [B,S,D]
    const float* W_in    = (const float*)d_in[2];   // [D,D]
    const float* W_out   = (const float*)d_in[3];   // [D,2D]

    float* attn_out  = (float*)d_out;                        // [T,B,D]
    float* align_out = (float*)d_out + (size_t)TT * BB * DD; // [T,B,S]

    // Workspace: all-bf16 operand pool (~342 MB)
    const long nBTD = (long)BB * TT * DD;
    const long nBSD = (long)BB * SS * DD;
    const long nBTS = (long)BB * TT * SS;
    u16* in_bf   = (u16*)d_ws;
    u16* ctx_bf  = in_bf  + nBTD;
    u16* win_bf  = ctx_bf + nBSD;
    u16* wout_bf = win_bf + (long)DD * DD;
    u16* h_bf    = wout_bf + (long)DD * 2 * DD;
    u16* algn_bf = h_bf   + nBTD;
    u16* c_bf    = algn_bf + nBTS;

    const dim3 blk(256);

    // 0) one-shot fp32 -> bf16 packs (keeps conversion out of GEMM loops)
    pack_bf16<<<dim3((unsigned)(nBTD / 4 / 256)), blk, 0, stream>>>(input, in_bf, nBTD);
    pack_bf16<<<dim3((unsigned)(nBSD / 4 / 256)), blk, 0, stream>>>(context, ctx_bf, nBSD);
    pack_bf16<<<dim3((unsigned)((long)DD * DD / 4 / 256)), blk, 0, stream>>>(W_in, win_bf, (long)DD * DD);
    pack_bf16<<<dim3((unsigned)((long)DD * 2 * DD / 4 / 256)), blk, 0, stream>>>(W_out, wout_bf, (long)DD * 2 * DD);

    const dim3 grd(DD / BNT, TT / BMT, BB);   // (8, 8, 32)

    // 1) h[b,t,e] = sum_d input[b,t,d] * W_in[e,d]  -> bf16
    gemm_wmma_bf16<false, 0><<<grd, blk, 0, stream>>>(
        in_bf, in_bf, DD, win_bf, h_bf,
        DD, DD,
        (long)TT * DD, DD,
        0,
        (long)TT * DD, DD);

    // 2) align[t,b,s] = sum_d h[b,t,d] * context[b,s,d] -> raw f32 scores
    //    written straight into d_out align region with [T,B,S] strides
    gemm_wmma_bf16<false, 1><<<grd, blk, 0, stream>>>(
        h_bf, h_bf, DD, ctx_bf, align_out,
        SS, DD,
        (long)TT * DD, DD,
        (long)SS * DD,
        (long)SS, (long)BB * SS);

    // 3) softmax over s in place + bf16 copy at [B,T,S]
    softmax_rows<<<dim3(BB * TT), blk, 0, stream>>>(align_out, algn_bf);

    // 4) c[b,t,d] = sum_s align_v[b,t,s] * context[b,s,d]  (B is [K,N]) -> bf16
    gemm_wmma_bf16<true, 0><<<grd, blk, 0, stream>>>(
        algn_bf, algn_bf, SS, ctx_bf, c_bf,
        DD, SS,
        (long)TT * SS, SS,
        (long)SS * DD,
        (long)TT * DD, DD);

    // 5) attn[t,b,d] = tanh( sum_f [c|input][b,t,f] * W_out[d,f] ), K=2D dual-A
    gemm_wmma_bf16<false, 2><<<grd, blk, 0, stream>>>(
        c_bf, in_bf, DD, wout_bf, attn_out,
        DD, 2 * DD,
        (long)TT * DD, DD,
        0,
        (long)DD, (long)BB * DD);
}
